// NoisyTopkRouter_8074538516585
// MI455X (gfx1250) — matrix operand order repaired and checked by reading, not compiled, not run on hardware
//
#include <hip/hip_runtime.h>
#include <math.h>

// ---------------------------------------------------------------------------
// NoisyTopkRouter for MI455X (gfx1250, wave32, WMMA).
//
// Shapes: B=8, T=4096 -> N_TOK=32768 tokens, D=192, H=4*D=768, E=6, TOP_K=2.
//
// Cost model: 9.7 GFLOP GEMM (x @ rw1[:D]) -> v_wmma_f32_16x16x32_f16
// (f32 accumulate); 151M exact-gelu evals -> branchless VALU erf fused
// against the WMMA accumulators so the (B,T,6,768) intermediate never
// touches memory. All working data is L2-resident (192MB L2), HBM traffic
// ~36MB total at 23.3 TB/s => the kernel is VALU-bound on the erf math,
// hence branch-free straight-line gelu.
// ---------------------------------------------------------------------------

typedef __attribute__((ext_vector_type(16))) _Float16 v16h;
typedef __attribute__((ext_vector_type(8)))  float    v8f;

#define N_TOK 32768
#define DIM   192
#define HID   768
#define NEXP  6
#define KCH   6      // K chunks of 32 (192 = 6*32)
#define NCH   48     // N chunks of 16 (768 = 48*16)

// Branchless erf (Abramowitz-Stegun 7.1.26, max abs err 1.5e-7):
//   erf(x) = sign(x) * (1 - poly(t) * exp(-x^2)),  t = 1/(1 + 0.3275911|x|)
// Uses raw v_rcp_f32 / v_exp_f32 (TRANS ops co-issue with the FMA stream);
// no EXEC-mask branches like libm's erff.
__device__ __forceinline__ float erf_approx(float x) {
  float ax = fabsf(x);
  float t  = __builtin_amdgcn_rcpf(fmaf(0.3275911f, ax, 1.0f));
  float p  = fmaf(1.061405429f, t, -1.453152027f);
  p = fmaf(p, t, 1.421413741f);
  p = fmaf(p, t, -0.284496736f);
  p = fmaf(p, t, 0.254829592f);
  p = p * t;
  // exp(-ax^2) = exp2(-ax^2 * log2(e))
  float e  = __builtin_amdgcn_exp2f(ax * ax * -1.4426950408889634f);
  float r  = fmaf(-p, e, 1.0f);
  return __builtin_copysignf(r, x);
}

__device__ __forceinline__ float gelu_exact(float v) {
  // jax.nn.gelu(approximate=False): 0.5*v*(1+erf(v/sqrt(2)))
  return 0.5f * v * (1.0f + erf_approx(v * 0.70710678118654752f));
}
__device__ __forceinline__ float softplus_f(float v) {
  // numerically stable log(1+exp(v))
  return fmaxf(v, 0.0f) + log1pf(expf(-fabsf(v)));
}

// ---------------------------------------------------------------------------
// Kernel 0: small precompute.
//   cTab[e][j] = (type_emb[expert_types[e]] @ rw1[D:3D])[j] + rb1[j]
//   w2m[j]    = rw2[j,:].mean()
//   meta[0]   = rb2.mean()
//   zero entropy partials + load counters (re-zeroed every launch so the
//   graph replay is deterministic).
// ---------------------------------------------------------------------------
__global__ __launch_bounds__(768) void k_setup(
    const float* __restrict__ type_emb, const int* __restrict__ expert_types,
    const float* __restrict__ rw1, const float* __restrict__ rb1,
    const float* __restrict__ rw2, const float* __restrict__ rb2,
    float* __restrict__ cTab, float* __restrict__ w2m, float* __restrict__ meta,
    float* __restrict__ entPartial, unsigned int* __restrict__ loadCnt) {
  const int j = threadIdx.x;            // 0..767
  // w2m
  float s = 0.0f;
  #pragma unroll
  for (int e = 0; e < NEXP; ++e) s += rw2[j * NEXP + e];
  w2m[j] = s * (1.0f / 6.0f);
  // cTab
  #pragma unroll
  for (int e = 0; e < NEXP; ++e) {
    const float* te = type_emb + expert_types[e] * (2 * DIM);
    float acc = rb1[j];
    for (int d = 0; d < 2 * DIM; ++d)
      acc = fmaf(te[d], rw1[(DIM + d) * HID + j], acc);
    cTab[e * HID + j] = acc;
  }
  if (j == 0) {
    float b = 0.0f;
    #pragma unroll
    for (int e = 0; e < NEXP; ++e) b += rb2[e];
    meta[0] = b * (1.0f / 6.0f);
  }
  if (j < 128) entPartial[j] = 0.0f;
  if (j < NEXP) loadCnt[j] = 0u;
}

// ---------------------------------------------------------------------------
// Kernel 1: pack wx = rw1[:D] (f32 row-major 192x768) into per-lane WMMA
// B-matrix layout, f16. Layout assumed for V_WMMA_F32_16X16X32_F16 B (32x16):
//   lane l: column N = l%16; lanes 0-15 hold K=k0+{0..7,16..23}, lanes 16-31
//   hold K=k0+{8..15,24..31} (same K order as the A operand).
// pB index = ((kc*48 + nc)*32 + lane)*16 + h   -> 147456 halves = 288 KB,
// L2-resident and reused by all 2048 token tiles.
// ---------------------------------------------------------------------------
__global__ __launch_bounds__(256) void k_packB(const float* __restrict__ rw1,
                                               _Float16* __restrict__ pB) {
  int tid = blockIdx.x * 256 + threadIdx.x;   // 576 blocks * 256 = 147456
  int h = tid & 15;
  int t = tid >> 4;
  int lane = t & 31; t >>= 5;
  int nc = t % NCH;
  int kc = t / NCH;
  int n = nc * 16 + (lane & 15);
  int k = kc * 32 + ((lane < 16) ? 0 : 8) + ((h < 8) ? h : h + 8);
  pB[tid] = (_Float16)rw1[k * HID + n];       // wx = rw1[:192] row-major
}

// ---------------------------------------------------------------------------
// Kernel 2: fused  logits = gelu(x@wx [+c_e]) @ w2m  via WMMA.
// One wave per 16-token tile (2048 tiles). For each of 48 N-chunks:
//   6x v_wmma_f32_16x16x32_f16 chained over K -> 16x16 f32 xpart tile,
//   then per element: 6 expert gelu-dot partial sums, finally a 16-lane
//   butterfly reduction over the N dimension.
// C/D layout: VGPR r, lanes 0-15 -> M=r, lanes 16-31 -> M=r+8; N = lane%16.
// A layout (16x32 f16): lane l -> row M=l%16; lanes<16 hold K={0..7,16..23},
// lanes>=16 hold K={8..15,24..31} (ISA 7.12.2).
// ---------------------------------------------------------------------------
__global__ __launch_bounds__(256) void k_logits(
    const float* __restrict__ x, const _Float16* __restrict__ pB,
    const float* __restrict__ cTab, const float* __restrict__ w2m,
    const float* __restrict__ meta, float* __restrict__ logits) {
  const int lane = threadIdx.x & 31;
  const int wave = threadIdx.x >> 5;
  const int tile = blockIdx.x * 8 + wave;          // 256 blocks * 8 waves = 2048
  const int tok0 = tile * 16;
  const int mrow = lane & 15;
  const float* xr = x + (size_t)(tok0 + mrow) * DIM;
  const int kbase = (lane < 16) ? 0 : 8;

  // Load + convert the A tiles for all 6 K-chunks (reused across 48 N-chunks).
  v16h a[KCH];
  #pragma unroll
  for (int kc = 0; kc < KCH; ++kc) {
    const float* p0 = xr + kc * 32 + kbase;
    float4 f0 = *(const float4*)(p0);
    float4 f1 = *(const float4*)(p0 + 4);
    float4 f2 = *(const float4*)(p0 + 16);
    float4 f3 = *(const float4*)(p0 + 20);
    v16h av;
    av[0] = (_Float16)f0.x;  av[1] = (_Float16)f0.y;
    av[2] = (_Float16)f0.z;  av[3] = (_Float16)f0.w;
    av[4] = (_Float16)f1.x;  av[5] = (_Float16)f1.y;
    av[6] = (_Float16)f1.z;  av[7] = (_Float16)f1.w;
    av[8] = (_Float16)f2.x;  av[9] = (_Float16)f2.y;
    av[10] = (_Float16)f2.z; av[11] = (_Float16)f2.w;
    av[12] = (_Float16)f3.x; av[13] = (_Float16)f3.y;
    av[14] = (_Float16)f3.z; av[15] = (_Float16)f3.w;
    a[kc] = av;
  }

  float lacc[8][NEXP];
  #pragma unroll
  for (int r = 0; r < 8; ++r)
    #pragma unroll
    for (int e = 0; e < NEXP; ++e) lacc[r][e] = 0.0f;

  const float w2bias = meta[0];

  for (int nc = 0; nc < NCH; ++nc) {
    v8f acc = {};
    #pragma unroll
    for (int kc = 0; kc < KCH; ++kc) {
      v16h b = *(const v16h*)(pB + (size_t)(((kc * NCH + nc) * 32) + lane) * 16);
      acc = __builtin_amdgcn_wmma_f32_16x16x32_f16(
          /*neg_a=*/false, a[kc], /*neg_b=*/false, b,
          /*c_mod=*/(short)0, acc, /*reuse_a=*/false, /*reuse_b=*/false);
    }
    const int n = nc * 16 + mrow;                 // this lane's hidden index
    const float wv = w2m[n];
    float cv[NEXP];
    #pragma unroll
    for (int e = 0; e < NEXP; ++e) cv[e] = cTab[e * HID + n];
    #pragma unroll
    for (int r = 0; r < 8; ++r) {
      float v = acc[r];
      #pragma unroll
      for (int e = 0; e < NEXP; ++e)
        lacc[r][e] = fmaf(gelu_exact(v + cv[e]), wv, lacc[r][e]);
    }
  }

  // Reduce over the 16 N-lanes inside each half-wave.
  #pragma unroll
  for (int m = 1; m < 16; m <<= 1)
    #pragma unroll
    for (int r = 0; r < 8; ++r)
      #pragma unroll
      for (int e = 0; e < NEXP; ++e)
        lacc[r][e] += __shfl_xor(lacc[r][e], m, 32);

  if (lane == 0 || lane == 16) {
    const int rbase = tok0 + ((lane == 16) ? 8 : 0);
    #pragma unroll
    for (int r = 0; r < 8; ++r)
      #pragma unroll
      for (int e = 0; e < NEXP; ++e)
        logits[(size_t)(rbase + r) * NEXP + e] = lacc[r][e] + w2bias;
  }
}

// ---------------------------------------------------------------------------
// Kernel 3: per-token noise MLP, noisy logits, top-2, sparse softmax,
// entropy / load-count reductions (deterministic: fixed LDS trees + integer
// atomics).
// ---------------------------------------------------------------------------
__global__ __launch_bounds__(256) void k_finalize(
    const float* __restrict__ x, const float* __restrict__ noise,
    const int* __restrict__ expert_types,
    const float* __restrict__ nw1, const float* __restrict__ nb1,
    const float* __restrict__ nw2, const float* __restrict__ nb2,
    const float* __restrict__ temperature, const float* __restrict__ logits,
    float* __restrict__ outP, float* __restrict__ outIdx,
    float* __restrict__ entPartial, unsigned int* __restrict__ loadCnt) {
  __shared__ float sEnt[256];
  __shared__ unsigned int sLoad[NEXP];
  if (threadIdx.x < NEXP) sLoad[threadIdx.x] = 0u;
  __syncthreads();

  const int t = blockIdx.x * 256 + threadIdx.x;   // 128 blocks -> 32768 tokens
  const float* xr = x + (size_t)t * DIM;

  // hidden = gelu(x @ nw1 + nb1)   (192 -> 12)
  float g[12];
  {
    float s[12];
    #pragma unroll
    for (int j = 0; j < 12; ++j) s[j] = nb1[j];
    for (int k = 0; k < DIM; ++k) {
      float xv = xr[k];
      #pragma unroll
      for (int j = 0; j < 12; ++j) s[j] = fmaf(xv, nw1[k * 12 + j], s[j]);
    }
    #pragma unroll
    for (int j = 0; j < 12; ++j) g[j] = gelu_exact(s[j]);
  }
  // nscale = softplus(softplus(hidden @ nw2 + nb2))   (12 -> 6)
  float nsc[NEXP];
  #pragma unroll
  for (int e = 0; e < NEXP; ++e) {
    float s = nb2[e];
    #pragma unroll
    for (int j = 0; j < 12; ++j) s = fmaf(g[j], nw2[j * NEXP + e], s);
    nsc[e] = softplus_f(softplus_f(s));
  }

  // temp = clip(temperature * 0.95**(4096//100), 0.05, 3.0)
  float temp = temperature[0] * 0.12851215678f;   // 0.95**40
  temp = fminf(fmaxf(temp, 0.05f), 3.0f);

  float noisy[NEXP];
  #pragma unroll
  for (int e = 0; e < NEXP; ++e) {
    float bonus = (expert_types[e] == 1) ? 0.3f : 0.0f;
    noisy[e] = logits[(size_t)t * NEXP + e]
             + temp * (noise[(size_t)t * NEXP + e] * nsc[e]) + bonus;
  }

  // top-2, JAX tie-breaking (lower index first on equal values)
  int i0 = 0; float v0 = noisy[0];
  #pragma unroll
  for (int e = 1; e < NEXP; ++e) if (noisy[e] > v0) { v0 = noisy[e]; i0 = e; }
  int i1 = -1; float v1 = -3.402823466e38f;
  #pragma unroll
  for (int e = 0; e < NEXP; ++e)
    if (e != i0 && noisy[e] > v1) { v1 = noisy[e]; i1 = e; }

  // 2-way softmax (v0 >= v1)
  float ex = expf(v1 - v0);
  float inv = 1.0f / (1.0f + ex);
  float p0 = inv, p1 = ex * inv;

  #pragma unroll
  for (int e = 0; e < NEXP; ++e)
    outP[(size_t)t * NEXP + e] = (e == i0) ? p0 : ((e == i1) ? p1 : 0.0f);
  outIdx[(size_t)t * 2 + 0] = (float)i0;
  outIdx[(size_t)t * 2 + 1] = (float)i1;

  float ent = -(p0 * logf(p0 + 1e-8f) + p1 * logf(p1 + 1e-8f));

  sEnt[threadIdx.x] = ent;
  atomicAdd(&sLoad[i0], 1u);
  atomicAdd(&sLoad[i1], 1u);
  __syncthreads();
  #pragma unroll
  for (int s = 128; s > 0; s >>= 1) {
    if (threadIdx.x < s) sEnt[threadIdx.x] += sEnt[threadIdx.x + s];
    __syncthreads();
  }
  if (threadIdx.x == 0) entPartial[blockIdx.x] = sEnt[0];
  if (threadIdx.x < NEXP) atomicAdd(&loadCnt[threadIdx.x], sLoad[threadIdx.x]);
}

// ---------------------------------------------------------------------------
// Kernel 4: aux_loss = 0.1*entropy_mean + 0.2*(std(load,ddof=1) + std(importance))
// std(importance) == 0 exactly: every token selects exactly TOP_K=2 experts,
// so importance[t] = 8*2/6 is the same for all t.
// ---------------------------------------------------------------------------
__global__ __launch_bounds__(128) void k_aux(
    const float* __restrict__ entPartial,
    const unsigned int* __restrict__ loadCnt, float* __restrict__ outAux) {
  __shared__ float s[128];
  s[threadIdx.x] = entPartial[threadIdx.x];
  __syncthreads();
  #pragma unroll
  for (int k = 64; k > 0; k >>= 1) {
    if (threadIdx.x < k) s[threadIdx.x] += s[threadIdx.x + k];
    __syncthreads();
  }
  if (threadIdx.x == 0) {
    float entMean = s[0] / (float)N_TOK;
    float c[NEXP], mean = 0.0f;
    #pragma unroll
    for (int e = 0; e < NEXP; ++e) { c[e] = (float)loadCnt[e]; mean += c[e]; }
    mean *= (1.0f / 6.0f);
    float var = 0.0f;
    #pragma unroll
    for (int e = 0; e < NEXP; ++e) { float d = c[e] - mean; var += d * d; }
    var *= (1.0f / 5.0f);                       // ddof=1
    outAux[0] = 0.1f * entMean + 0.2f * (sqrtf(var) + 0.0f);
  }
}

// ---------------------------------------------------------------------------
extern "C" void kernel_launch(void* const* d_in, const int* in_sizes, int n_in,
                              void* d_out, int out_size, void* d_ws, size_t ws_size,
                              hipStream_t stream) {
  const float* x      = (const float*)d_in[0];
  const float* noise  = (const float*)d_in[1];
  const int*   etypes = (const int*)d_in[2];
  const float* temb   = (const float*)d_in[3];
  const float* nw1    = (const float*)d_in[4];
  const float* nb1    = (const float*)d_in[5];
  const float* nw2    = (const float*)d_in[6];
  const float* nb2    = (const float*)d_in[7];
  const float* rw1    = (const float*)d_in[8];
  const float* rb1    = (const float*)d_in[9];
  const float* rw2    = (const float*)d_in[10];
  const float* rb2    = (const float*)d_in[11];
  const float* temper = (const float*)d_in[12];

  float* out = (float*)d_out;
  float* outP   = out;                    // 32768*6
  float* outIdx = out + 196608;           // 32768*2 (int indices as floats)
  float* outAux = out + 262144;           // scalar

  char* ws = (char*)d_ws;
  float*        logits     = (float*)(ws);                 // 786432 B
  _Float16*     pB         = (_Float16*)(ws + 786432);     // 294912 B
  float*        cTab       = (float*)(ws + 1081344);       // 18432 B
  float*        w2m        = (float*)(ws + 1099776);       // 3072 B
  float*        meta       = (float*)(ws + 1102848);       // 64 B
  float*        entPartial = (float*)(ws + 1102912);       // 512 B
  unsigned int* loadCnt    = (unsigned int*)(ws + 1103424);// 24 B

  k_setup<<<1, 768, 0, stream>>>(temb, etypes, rw1, rb1, rw2, rb2,
                                 cTab, w2m, meta, entPartial, loadCnt);
  k_packB<<<576, 256, 0, stream>>>(rw1, pB);
  k_logits<<<256, 256, 0, stream>>>(x, pB, cTab, w2m, meta, logits);
  k_finalize<<<128, 256, 0, stream>>>(x, noise, etypes, nw1, nb1, nw2, nb2,
                                      temper, logits, outP, outIdx,
                                      entPartial, loadCnt);
  k_aux<<<1, 128, 0, stream>>>(entPartial, loadCnt, outAux);
}
